// Caterpillar_69415261438746
// MI455X (gfx1250) — compile-verified, matrix operand order checked by reading, not run on hardware
//
#include <hip/hip_runtime.h>
#include <hip/hip_bf16.h>
#include <math.h>
#include <stdint.h>

// Problem constants (from reference)
#define NN 2
#define TT 4096
#define DD 1024
#define HH 8
#define EE 4
#define LL 32
#define DKV 64
#define MROWS (NN * TT)       // 8192
#define RECP 4128             // L + T
#define RECPV 4160            // padded so WMMA K-pad reads stay in bounds

typedef __attribute__((ext_vector_type(16))) __bf16 v16bf;
typedef __attribute__((ext_vector_type(8)))  __bf16 v8bf;
typedef __attribute__((ext_vector_type(8)))  float  v8f;
typedef __attribute__((ext_vector_type(4)))  int    v4i;

// ---------------------------------------------------------------------------
// CDNA5 async global->LDS copy (ASYNCcnt path), with asm fallback.
// Builtin signature (from hipcc diagnostic): param0 = v4i AS1*, param1 = AS3.
// AS3 pointer value == LDS byte offset (flat aperture truncates to addr[31:0]).
// ---------------------------------------------------------------------------
__device__ __forceinline__ void async_b128(const void* g, void* s) {
#if defined(__AMDGCN__)
#if __has_builtin(__builtin_amdgcn_global_load_async_to_lds_b128)
    __builtin_amdgcn_global_load_async_to_lds_b128(
        (__attribute__((address_space(1))) v4i*)(unsigned long long)(uintptr_t)g,
        (__attribute__((address_space(3))) v4i*)(unsigned)(uintptr_t)s,
        0, 0);
#else
    unsigned so = (unsigned)(uintptr_t)s;
    asm volatile("global_load_async_to_lds_b128 %0, %1, off"
                 :: "v"(so), "v"(g) : "memory");
#endif
#endif
}

__device__ __forceinline__ void wait_async3() {
#if defined(__AMDGCN__)
#if __has_builtin(__builtin_amdgcn_s_wait_asynccnt)
    __builtin_amdgcn_s_wait_asynccnt(3);
#else
    asm volatile("s_wait_asynccnt 0x3" ::: "memory");
#endif
#endif
}

__device__ __forceinline__ void wait_async0() {
#if defined(__AMDGCN__)
#if __has_builtin(__builtin_amdgcn_s_wait_asynccnt)
    __builtin_amdgcn_s_wait_asynccnt(0);
#else
    asm volatile("s_wait_asynccnt 0x0" ::: "memory");
#endif
#endif
}

// ---------------------------------------------------------------------------
// 16-bit WMMA A/B fragment loader (row-major source, works for global or LDS).
// lane&15 selects the M-row (A) / N-column-as-Bt-row (B); khalf=(lane>>4)*8.
// halves 0..7  -> K = k0 + khalf + i ; halves 8..15 -> K = k0 + 16 + khalf + i
// ---------------------------------------------------------------------------
__device__ __forceinline__ v16bf load_frag(const __bf16* rowptr, int k0, int hk) {
    v8bf lo = *(const v8bf*)(rowptr + k0 + hk);
    v8bf hi = *(const v8bf*)(rowptr + k0 + 16 + hk);
    v16bf r;
#pragma unroll
    for (int i = 0; i < 8; ++i) { r[i] = lo[i]; r[i + 8] = hi[i]; }
    return r;
}

__device__ __forceinline__ v8f wmma_bf16(v16bf a, v16bf b, v8f c) {
    return __builtin_amdgcn_wmma_f32_16x16x32_bf16(false, a, false, b, (short)0, c,
                                                   false, false);
}

// ---------------------------------------------------------------------------
// f32 -> bf16 conversion
// ---------------------------------------------------------------------------
__global__ void cvt_kernel(const float* __restrict__ s, __bf16* __restrict__ d, int n) {
    int i = blockIdx.x * blockDim.x + threadIdx.x;
    if (i < n) d[i] = (__bf16)s[i];
}

// w_O (512 x 1024) -> w_OT (1024 x 512) bf16
__global__ void wot_kernel(const float* __restrict__ wO, __bf16* __restrict__ wOT) {
    int i = blockIdx.x * blockDim.x + threadIdx.x;   // i = j*512 + c
    if (i >= 1024 * 512) return;
    int c = i & 511;
    int j = i >> 9;
    wOT[i] = (__bf16)wO[(size_t)c * 1024 + j];
}

// ---------------------------------------------------------------------------
// Tiled GEMM: Out[M x Cout] = A[M x K] * B^T, B row-major (Cout x K), bf16.
// Block = 256 threads (8 waves); block tile 128(M) x 64(N); k-step 32.
// A(128x32) + B(64x32) staged in LDS by async b128 copies, double buffered:
// issue next k-step, s_wait_asynccnt 3 (current group done, next in flight),
// barrier, 4 WMMAs per wave from ds_load_b128 fragments.
// ---------------------------------------------------------------------------
template <int OUT_BF16>
__global__ __launch_bounds__(256) void gemm_tile(const __bf16* __restrict__ A,
                                                 const __bf16* __restrict__ B,
                                                 void* __restrict__ Out, int M,
                                                 int Cout, int K, int ldo) {
    __shared__ __bf16 smA[2][128 * 32];   // 2 x 8 KB
    __shared__ __bf16 smB[2][64 * 32];    // 2 x 4 KB

    const int nt64cnt = (Cout + 63) >> 6;
    const int mt = blockIdx.x / nt64cnt;      // 128-row tile
    const int nt = blockIdx.x % nt64cnt;      // 64-col tile
    const int tid = threadIdx.x;
    const int wv = tid >> 5;
    const int lane = tid & 31;
    const int m = lane & 15;
    const int hk = (lane >> 4) << 3;
    const int rbase = (lane >> 4) << 3;

    // issue one k-step's async copies: 2 A chunks + 1 B chunk per thread
    auto issue = [&](int buf, int kb) {
#pragma unroll
        for (int q = 0; q < 2; ++q) {
            int c = tid + q * 256;            // 512 x 16B chunks of A tile
            int row = c >> 2, part = c & 3;
            const char* g = (const char*)A +
                ((size_t)(mt * 128 + row) * K + kb) * 2 + part * 16;
            char* s = (char*)&smA[buf][0] + row * 64 + part * 16;
            async_b128(g, s);
        }
        {
            int row = tid >> 2, part = tid & 3;   // 256 x 16B chunks of B tile
            int br = nt * 64 + row;
            if (br >= Cout) br = Cout - 1;        // clamped read, store guarded
            const char* g = (const char*)B + ((size_t)br * K + kb) * 2 + part * 16;
            char* s = (char*)&smB[buf][0] + row * 64 + part * 16;
            async_b128(g, s);
        }
    };

    v8f acc[4];
#pragma unroll
    for (int j = 0; j < 4; ++j)
#pragma unroll
        for (int i = 0; i < 8; ++i) acc[j][i] = 0.f;

    const int nk = K >> 5;
    issue(0, 0);
    for (int i = 0; i < nk; ++i) {
        const int buf = i & 1;
        if (i + 1 < nk) {
            issue(buf ^ 1, (i + 1) << 5);
            wait_async3();                     // group i complete, group i+1 inflight
        } else {
            wait_async0();
        }
        __syncthreads();

        const __bf16* arow = &smA[buf][(wv * 16 + m) * 32];
        v16bf af = load_frag(arow, 0, hk);
#pragma unroll
        for (int j = 0; j < 4; ++j) {
            const __bf16* brow = &smB[buf][(j * 16 + m) * 32];
            v16bf bf_ = load_frag(brow, 0, hk);
            acc[j] = wmma_bf16(af, bf_, acc[j]);
        }
        __syncthreads();                       // everyone done before overwrite
    }

#pragma unroll
    for (int j = 0; j < 4; ++j) {
        int col = nt * 64 + j * 16 + m;
        if (col >= Cout) continue;
#pragma unroll
        for (int r = 0; r < 8; ++r) {
            int row = mt * 128 + wv * 16 + rbase + r;
            float v = acc[j][r];
            if (OUT_BF16) ((__bf16*)Out)[(size_t)row * ldo + col] = (__bf16)v;
            else          ((float*)Out)[(size_t)row * ldo + col] = v;
        }
    }
}

// ---------------------------------------------------------------------------
// Gate: G = sigmoid(Graw + b), A = 1 - min(sum_h G, 1), gK/gV = sum_h G*K/V
// ---------------------------------------------------------------------------
__global__ void gate_kernel(const float* __restrict__ Graw, const float* __restrict__ bG,
                            const __bf16* __restrict__ Kb, const __bf16* __restrict__ Vb,
                            float* __restrict__ Aout, float* __restrict__ gK,
                            float* __restrict__ gV) {
    int idx = blockIdx.x * blockDim.x + threadIdx.x;   // ((r*4)+e)*64 + d
    int d = idx & 63;
    int e = (idx >> 6) & 3;
    int r = idx >> 8;
    if (r >= MROWS) return;

    float gsum = 0.f, gk = 0.f, gv = 0.f;
#pragma unroll
    for (int h = 0; h < HH; ++h) {
        float g = Graw[(size_t)r * 32 + h * 4 + e] + bG[h * 4 + e];
        g = 1.f / (1.f + __expf(-g));
        gsum += g;
        gk += g * (float)Kb[(size_t)r * 512 + h * 64 + d];
        gv += g * (float)Vb[(size_t)r * 512 + h * 64 + d];
    }
    int n = r >> 12;
    int t = r & 4095;
    size_t o = ((size_t)(n * 4 + e) * TT + t) * 64 + d;
    gK[o] = gk;
    gV[o] = gv;
    if (d == 0) Aout[(size_t)(n * 4 + e) * TT + t] = 1.f - fminf(gsum, 1.f);
}

// ---------------------------------------------------------------------------
// Scan: per (n,e,chunk l,d): state = A*state + gX over 128 steps.
// recK (n,e,RECP,64) bf16, recVT (n,e,64,RECPV) bf16; recK[p<32]=init rows.
// ---------------------------------------------------------------------------
__global__ void scan_kernel(const float* __restrict__ A, const float* __restrict__ gK,
                            const float* __restrict__ gV, const float* __restrict__ iK,
                            const float* __restrict__ iV, __bf16* __restrict__ recK,
                            __bf16* __restrict__ recVT) {
    int idx = blockIdx.x * blockDim.x + threadIdx.x;   // ((ne)*32 + l)*64 + d
    int d = idx & 63;
    int l = (idx >> 6) & 31;
    int ne = idx >> 11;
    if (ne >= NN * EE) return;
    int e = ne & 3;

    float sK = iK[((size_t)e * 32 + l) * 64 + d];
    float sV = iV[((size_t)e * 32 + l) * 64 + d];

    __bf16* RK  = recK + (size_t)ne * RECP * 64;
    __bf16* RVT = recVT + ((size_t)ne * 64 + d) * RECPV;

    RK[(size_t)l * 64 + d] = (__bf16)sK;
    RVT[l] = (__bf16)sV;
    RVT[RECP + l] = (__bf16)0.f;                       // zero the K-pad tail

    const float* Ap  = A  + (size_t)ne * TT;
    const float* gKp = gK + (size_t)ne * TT * 64;
    const float* gVp = gV + (size_t)ne * TT * 64;

    for (int c = 0; c < 128; ++c) {
        int t = l * 128 + c;
        float a = Ap[t];
        sK = a * sK + gKp[(size_t)t * 64 + d];
        sV = a * sV + gVp[(size_t)t * 64 + d];
        RK[(size_t)(32 + t) * 64 + d] = (__bf16)sK;
        RVT[32 + t] = (__bf16)sV;
    }
}

// ---------------------------------------------------------------------------
// Attention: one wave per (n, h, 16-query tile).
// Query t attends recK positions p = t+1 .. t+32 (band): 3 aligned key tiles
// x 4 experts, banded WMMA + mask; softmax over 4x64 LDS cols (pads -1e30);
// PV: probs (bf16 A-frag from LDS) x recVT rows.
// ---------------------------------------------------------------------------
__global__ __launch_bounds__(32) void attn_kernel(const __bf16* __restrict__ Qb,
                                                  const __bf16* __restrict__ recK,
                                                  const __bf16* __restrict__ recVT,
                                                  __bf16* __restrict__ Y) {
    const int t0 = blockIdx.x * 16;
    const int h = blockIdx.y;
    const int n = blockIdx.z;
    const int lane = threadIdx.x;
    const int m = lane & 15;
    const int hk = (lane >> 4) << 3;
    const int rbase = (lane >> 4) << 3;

    __shared__ float sm[16][4][64];
    for (int i = lane; i < 16 * 4 * 64; i += 32) ((float*)sm)[i] = -1e30f;
    __syncthreads();

    const __bf16* qrow = Qb + (size_t)(n * TT + t0 + m) * 512 + h * 64;
    v16bf qa0 = load_frag(qrow, 0, hk);
    v16bf qa1 = load_frag(qrow, 32, hk);

    for (int e = 0; e < EE; ++e) {
        const __bf16* RK = recK + (size_t)((n * 4 + e) * RECP) * 64;
#pragma unroll
        for (int wt = 0; wt < 3; ++wt) {
            int p0 = t0 + wt * 16;
            const __bf16* krow = RK + (size_t)(p0 + m) * 64;
            v8f c;
#pragma unroll
            for (int i = 0; i < 8; ++i) c[i] = 0.f;
            c = wmma_bf16(qa0, load_frag(krow, 0, hk), c);
            c = wmma_bf16(qa1, load_frag(krow, 32, hk), c);
#pragma unroll
            for (int r = 0; r < 8; ++r) {
                int mm = rbase + r;          // query row within tile
                int k = wt * 16 + m;         // band column = p - t0
                int rel = k - mm;            // p - t
                float s = (rel >= 1 && rel <= 32) ? c[r] * 0.125f : -1e30f;
                sm[mm][e][k] = s;
            }
        }
    }
    __syncthreads();

    if (lane < 16) {
        float* row = &sm[lane][0][0];
        float mx = -1e30f;
        for (int i = 0; i < 256; ++i) mx = fmaxf(mx, row[i]);
        float sum = 0.f;
        for (int i = 0; i < 256; ++i) {
            float ex = __expf(row[i] - mx);
            row[i] = ex;
            sum += ex;
        }
        float inv = 1.f / sum;
        for (int i = 0; i < 256; ++i) row[i] *= inv;
    }
    __syncthreads();

    v8f acc[4];
#pragma unroll
    for (int j = 0; j < 4; ++j)
#pragma unroll
        for (int i = 0; i < 8; ++i) acc[j][i] = 0.f;

    for (int e = 0; e < EE; ++e) {
#pragma unroll
        for (int ks = 0; ks < 2; ++ks) {
            v16bf pa;
#pragma unroll
            for (int i = 0; i < 8; ++i) {
                pa[i]     = (__bf16)sm[m][e][ks * 32 + hk + i];
                pa[i + 8] = (__bf16)sm[m][e][ks * 32 + 16 + hk + i];
            }
#pragma unroll
            for (int j = 0; j < 4; ++j) {
                const __bf16* vrow =
                    recVT + ((size_t)((n * 4 + e) * 64 + j * 16 + m)) * RECPV + t0;
                v16bf vb = load_frag(vrow, ks * 32, hk);
                acc[j] = wmma_bf16(pa, vb, acc[j]);
            }
        }
    }

#pragma unroll
    for (int j = 0; j < 4; ++j)
#pragma unroll
        for (int r = 0; r < 8; ++r)
            Y[(size_t)(n * TT + t0 + rbase + r) * 512 + h * 64 + j * 16 + m] =
                (__bf16)acc[j][r];
}

// ---------------------------------------------------------------------------
// Host launcher
// ---------------------------------------------------------------------------
static size_t carve(size_t* off, size_t bytes) {
    size_t o = *off;
    *off = (o + bytes + 255) & ~(size_t)255;
    return o;
}

extern "C" void kernel_launch(void* const* d_in, const int* in_sizes, int n_in,
                              void* d_out, int out_size, void* d_ws, size_t ws_size,
                              hipStream_t stream) {
    const float* X   = (const float*)d_in[0];
    const float* wG  = (const float*)d_in[1];
    const float* bG  = (const float*)d_in[2];
    const float* wK  = (const float*)d_in[3];
    const float* wV  = (const float*)d_in[4];
    const float* wQ  = (const float*)d_in[5];
    const float* wO  = (const float*)d_in[6];
    const float* iK  = (const float*)d_in[7];
    const float* iV  = (const float*)d_in[8];

    char* ws = (char*)d_ws;
    size_t off = 0;
    __bf16* Xh   = (__bf16*)(ws + carve(&off, (size_t)MROWS * DD * 2));
    __bf16* wGh  = (__bf16*)(ws + carve(&off, (size_t)32 * DD * 2));
    __bf16* wKh  = (__bf16*)(ws + carve(&off, (size_t)512 * DD * 2));
    __bf16* wVh  = (__bf16*)(ws + carve(&off, (size_t)512 * DD * 2));
    __bf16* wQh  = (__bf16*)(ws + carve(&off, (size_t)512 * DD * 2));
    __bf16* wOT  = (__bf16*)(ws + carve(&off, (size_t)1024 * 512 * 2));
    float*  Graw = (float*)(ws + carve(&off, (size_t)MROWS * 32 * 4));
    __bf16* Kb   = (__bf16*)(ws + carve(&off, (size_t)MROWS * 512 * 2));
    __bf16* Vb   = (__bf16*)(ws + carve(&off, (size_t)MROWS * 512 * 2));
    __bf16* Qb   = (__bf16*)(ws + carve(&off, (size_t)MROWS * 512 * 2));
    float*  Abuf = (float*)(ws + carve(&off, (size_t)NN * EE * TT * 4));
    float*  gK   = (float*)(ws + carve(&off, (size_t)NN * EE * TT * 64 * 4));
    float*  gV   = (float*)(ws + carve(&off, (size_t)NN * EE * TT * 64 * 4));
    __bf16* recK = (__bf16*)(ws + carve(&off, (size_t)NN * EE * RECP * 64 * 2));
    __bf16* recVT= (__bf16*)(ws + carve(&off, (size_t)NN * EE * 64 * RECPV * 2));
    __bf16* Yf   = (__bf16*)(ws + carve(&off, (size_t)MROWS * 512 * 2));

    // 1. conversions
    cvt_kernel<<<(MROWS * DD + 255) / 256, 256, 0, stream>>>(X, Xh, MROWS * DD);
    cvt_kernel<<<(32 * DD + 255) / 256, 256, 0, stream>>>(wG, wGh, 32 * DD);
    cvt_kernel<<<(512 * DD + 255) / 256, 256, 0, stream>>>(wK, wKh, 512 * DD);
    cvt_kernel<<<(512 * DD + 255) / 256, 256, 0, stream>>>(wV, wVh, 512 * DD);
    cvt_kernel<<<(512 * DD + 255) / 256, 256, 0, stream>>>(wQ, wQh, 512 * DD);
    wot_kernel<<<(1024 * 512 + 255) / 256, 256, 0, stream>>>(wO, wOT);

    // 2. projection GEMMs: blocks = (M/128) * ceil(Cout/64)
    gemm_tile<0><<<(MROWS / 128) * 1, 256, 0, stream>>>(Xh, wGh, Graw, MROWS, 32, DD, 32);
    gemm_tile<1><<<(MROWS / 128) * 8, 256, 0, stream>>>(Xh, wKh, Kb, MROWS, 512, DD, 512);
    gemm_tile<1><<<(MROWS / 128) * 8, 256, 0, stream>>>(Xh, wVh, Vb, MROWS, 512, DD, 512);
    gemm_tile<1><<<(MROWS / 128) * 8, 256, 0, stream>>>(Xh, wQh, Qb, MROWS, 512, DD, 512);

    // 3. gating
    gate_kernel<<<(MROWS * EE * 64 + 255) / 256, 256, 0, stream>>>(Graw, bG, Kb, Vb,
                                                                   Abuf, gK, gV);

    // 4. scan
    scan_kernel<<<(NN * EE * LL * 64 + 255) / 256, 256, 0, stream>>>(Abuf, gK, gV, iK,
                                                                     iV, recK, recVT);

    // 5. attention
    {
        dim3 grid(TT / 16, HH, NN);
        attn_kernel<<<grid, 32, 0, stream>>>(Qb, recK, recVT, Yf);
    }

    // 6. output projection -> f32 d_out
    gemm_tile<0><<<(MROWS / 128) * 16, 256, 0, stream>>>(Yf, wOT, (float*)d_out, MROWS,
                                                         1024, 512, 1024);
}
